// GCN_2834678415990
// MI455X (gfx1250) — matrix operand order repaired and checked by reading, not compiled, not run on hardware
//
#include <hip/hip_runtime.h>
#include <hip/hip_bf16.h>
#include <stdint.h>

#define NNODES 100000
#define NEDGES 800000
#define NGRAPHS 2048
#define EMBD 300
#define FEATD 256
#define NLAYERS 5
#define LDK 320               // padded feature stride (multiple of 32)
#define CHUNKS 75             // 300/4 float4 chunks per node row
#define EPSV 1e-5f

typedef unsigned short u16;
typedef __attribute__((ext_vector_type(16))) __bf16 v16bf;
typedef __attribute__((ext_vector_type(8)))  float  v8f;

__device__ __forceinline__ u16 f2bf(float f) {
  uint32_t x = __float_as_uint(f);
  uint32_t r = x + 0x7FFFu + ((x >> 16) & 1u);   // round-to-nearest-even
  return (u16)(r >> 16);
}

__device__ __forceinline__ void atomAddF(float* p, float v) {
  __hip_atomic_fetch_add(p, v, __ATOMIC_RELAXED, __HIP_MEMORY_SCOPE_AGENT);
}

// ---------------- utility ----------------
__global__ void zero_f32(float* p, int n) {
  int i = blockIdx.x * blockDim.x + threadIdx.x;
  if (i < n) p[i] = 0.f;
}

// ---------------- degree / norm ----------------
__global__ void deg_accum(const int* col, float* deg) {
  int e = blockIdx.x * blockDim.x + threadIdx.x;
  if (e < NEDGES) atomAddF(&deg[col[e]], 1.f);
}
__global__ void make_dinv(float* deg) {
  int n = blockIdx.x * blockDim.x + threadIdx.x;
  if (n < NNODES) deg[n] = rsqrtf(deg[n] + 1.f);
}
__global__ void make_norm(const int* row, const int* col, const float* dinv, float* norm) {
  int e = blockIdx.x * blockDim.x + threadIdx.x;
  if (e < NEDGES) norm[e] = dinv[row[e]] * dinv[col[e]];
}

// ---------------- embedding -> bf16 activations (padded) ----------------
__global__ void embed(const int* x, const float* emb1, const float* emb2, u16* hbf) {
  int idx = blockIdx.x * blockDim.x + threadIdx.x;
  if (idx >= NNODES * LDK) return;
  int n = idx / LDK, j = idx - n * LDK;
  float v = 0.f;
  if (j < EMBD) v = emb1[x[2 * n] * EMBD + j] + emb2[x[2 * n + 1] * EMBD + j];
  hbf[idx] = f2bf(v);
}

// ---------------- weight transpose + bf16 convert: Wt[n][k] = W[k][n] ----------------
__global__ void convert_wT(const float* W, u16* Wt, int K, int Ncols, int Nrows) {
  int idx = blockIdx.x * blockDim.x + threadIdx.x;
  if (idx >= Nrows * LDK) return;
  int n = idx / LDK, k = idx - n * LDK;
  float v = (k < K && n < Ncols) ? W[k * Ncols + n] : 0.f;
  Wt[idx] = f2bf(v);
}

// ---------------- WMMA bf16 GEMM: C[m][n] = sum_k A[m][k]*B[n][k] (+bias) ----------------
// A: [M][LDK] bf16 row-major. B: [Nrows][LDK] bf16 (pre-transposed weights).
// Each wave computes a 16x64 output strip: 4 accumulators, A-fragment reused 4x.
__global__ void wmma_gemm(const u16* __restrict__ A, const u16* __restrict__ B,
                          float* __restrict__ C, const float* __restrict__ bias,
                          int Mtiles, int nStrips, int ldc) {
  int wave = threadIdx.x >> 5;
  int gw = blockIdx.x * (blockDim.x >> 5) + wave;
  if (gw >= Mtiles * nStrips) return;          // wave-uniform exit (EXEC stays all-1s)
  int mt = gw / nStrips, ns = gw - mt * nStrips;
  int lane = threadIdx.x & 31;
  int lm = lane & 15;                          // row-in-A-tile / col-in-B-tile
  int kh = lane >> 4;                          // K-half selector
  int m0 = mt * 16, n0 = ns * 64;

  const u16* arow = A + (size_t)(m0 + lm) * LDK + kh * 8;   // A frag: K in {kh*8..+7, 16+kh*8..+7}
  v8f zero = {0.f, 0.f, 0.f, 0.f, 0.f, 0.f, 0.f, 0.f};
  v8f acc[4] = {zero, zero, zero, zero};

  for (int k0 = 0; k0 < LDK; k0 += 32) {
    union { uint4 q[2]; v16bf v; } af;
    af.q[0] = *reinterpret_cast<const uint4*>(arow + k0);
    af.q[1] = *reinterpret_cast<const uint4*>(arow + k0 + 16);
#pragma unroll
    for (int t = 0; t < 4; ++t) {
      // B frag: lane holds column n0+t*16+lm, K in {kh*8..+7, 16+kh*8..+7}
      const u16* brow = B + (size_t)(n0 + t * 16 + lm) * LDK + kh * 8 + k0;
      union { uint4 q[2]; v16bf v; } bfu;
      bfu.q[0] = *reinterpret_cast<const uint4*>(brow);
      bfu.q[1] = *reinterpret_cast<const uint4*>(brow + 16);
      acc[t] = __builtin_amdgcn_wmma_f32_16x16x32_bf16(
          false, af.v, false, bfu.v, (short)0, acc[t], false, false);
    }
  }
#pragma unroll
  for (int t = 0; t < 4; ++t) {
    int n = n0 + t * 16 + lm;
    float badd = bias ? bias[n] : 0.f;
#pragma unroll
    for (int r = 0; r < 8; ++r) {
      int m = m0 + r + 8 * kh;                 // C/D layout: VGPR r -> M=r (lanes<16) / 8+r
      C[(size_t)m * ldc + n] = acc[t][r] + badd;
    }
  }
}

// ---------------- self-loop + bias: agg = dinv^2 * hw + b ----------------
__global__ void selfloop_bias(const float* hw, const float* dinv, const float* b, float* agg) {
  int idx = blockIdx.x * blockDim.x + threadIdx.x;
  if (idx >= NNODES * LDK) return;
  int n = idx / LDK, j = idx - n * LDK;
  float d = dinv[n];
  float bv = (j < EMBD) ? b[j] : 0.f;
  agg[idx] = d * d * hw[idx] + bv;
}

// ---------------- edge scatter: agg[col] += norm[e] * hw[row] ----------------
__global__ void edge_scatter(const int* __restrict__ row, const int* __restrict__ col,
                             const float* __restrict__ norm, const float* __restrict__ hw,
                             float* __restrict__ agg) {
  int idx = blockIdx.x * blockDim.x + threadIdx.x;
  if (idx >= NEDGES * CHUNKS) return;
  int e = idx / CHUNKS, c = idx - e * CHUNKS;
  float nm = norm[e];
  const float4 v = *reinterpret_cast<const float4*>(hw + (size_t)row[e] * LDK + c * 4);
  float* dst = agg + (size_t)col[e] * LDK + c * 4;
  atomAddF(dst + 0, nm * v.x);
  atomAddF(dst + 1, nm * v.y);
  atomAddF(dst + 2, nm * v.z);
  atomAddF(dst + 3, nm * v.w);
}

// ---------------- batch-norm statistics ----------------
__global__ void bn_stats(const float* __restrict__ h, float* sums, float* sumsq) {
  int j = threadIdx.x;                         // blockDim.x == LDK
  float s = 0.f, s2 = 0.f;
  for (int n = blockIdx.x; n < NNODES; n += gridDim.x) {
    float v = h[(size_t)n * LDK + j];
    s += v; s2 += v * v;
  }
  atomAddF(&sums[j], s);
  atomAddF(&sumsq[j], s2);
}

// ---------------- batch-norm apply (+relu) -> bf16 (and optional f32) ----------------
__global__ void bn_apply(const float* __restrict__ agg, const float* sums, const float* sumsq,
                         const float* gamma, const float* beta,
                         u16* __restrict__ hbf, float* __restrict__ hout,
                         int do_relu, int write_f32) {
  int idx = blockIdx.x * blockDim.x + threadIdx.x;
  if (idx >= NNODES * LDK) return;
  int n = idx / LDK, j = idx - n * LDK; (void)n;
  float y = 0.f;
  if (j < EMBD) {
    const float invN = 1.0f / (float)NNODES;
    float mean = sums[j] * invN;
    float var = sumsq[j] * invN - mean * mean;
    var = fmaxf(var, 0.f);
    float inv = rsqrtf(var + EPSV);
    y = gamma[j] * (agg[idx] - mean) * inv + beta[j];
    if (do_relu) y = fmaxf(y, 0.f);
  }
  hbf[idx] = f2bf(y);
  if (write_f32) hout[idx] = y;
}

// ---------------- pooling ----------------
__global__ void count_nodes(const int* batch, float* counts) {
  int n = blockIdx.x * blockDim.x + threadIdx.x;
  if (n < NNODES) atomAddF(&counts[batch[n]], 1.f);
}
__global__ void pool_acc(const int* __restrict__ batch, const float* __restrict__ h,
                         float* __restrict__ pooled) {
  int idx = blockIdx.x * blockDim.x + threadIdx.x;
  if (idx >= NNODES * CHUNKS) return;
  int n = idx / CHUNKS, c = idx - n * CHUNKS;
  const float4 v = *reinterpret_cast<const float4*>(h + (size_t)n * LDK + c * 4);
  float* dst = pooled + (size_t)batch[n] * LDK + c * 4;
  atomAddF(dst + 0, v.x);
  atomAddF(dst + 1, v.y);
  atomAddF(dst + 2, v.z);
  atomAddF(dst + 3, v.w);
}
__global__ void pool_div(const float* pooled, const float* counts, u16* pbf) {
  int idx = blockIdx.x * blockDim.x + threadIdx.x;
  if (idx >= NGRAPHS * LDK) return;
  int g = idx / LDK, j = idx - g * LDK;
  float cnt = fmaxf(counts[g], 1.f);
  float v = (j < EMBD) ? pooled[idx] / cnt : 0.f;
  pbf[idx] = f2bf(v);
}

static inline int cdiv(long long a, long long b) { return (int)((a + b - 1) / b); }

extern "C" void kernel_launch(void* const* d_in, const int* in_sizes, int n_in,
                              void* d_out, int out_size, void* d_ws, size_t ws_size,
                              hipStream_t stream) {
  (void)in_sizes; (void)n_in; (void)out_size; (void)ws_size;
  const int*   x      = (const int*)d_in[0];
  const int*   eidx   = (const int*)d_in[1];
  const int*   batch  = (const int*)d_in[2];
  const float* emb1   = (const float*)d_in[3];
  const float* emb2   = (const float*)d_in[4];
  const float* Ws     = (const float*)d_in[5];
  const float* bs     = (const float*)d_in[6];
  const float* gammas = (const float*)d_in[7];
  const float* betas  = (const float*)d_in[8];
  const float* Wf     = (const float*)d_in[9];
  const float* bfb    = (const float*)d_in[10];
  float* out = (float*)d_out;
  const int* row = eidx;
  const int* col = eidx + NEDGES;

  char* p = (char*)d_ws;
  auto take = [&](size_t bytes) -> char* {
    char* r = p; p += (bytes + 255) & ~(size_t)255; return r;
  };
  float* bufA  = (float*)take((size_t)NNODES * LDK * 4);   // hw / BN output
  float* bufB  = (float*)take((size_t)NNODES * LDK * 4);   // agg
  u16*   hbf   = (u16*)  take((size_t)NNODES * LDK * 2);   // bf16 activations
  u16*   wbf   = (u16*)  take((size_t)LDK * LDK * 2);      // transposed layer weight
  u16*   wfbf  = (u16*)  take((size_t)FEATD * LDK * 2);    // transposed final weight
  float* dinv  = (float*)take((size_t)NNODES * 4);
  float* norm  = (float*)take((size_t)NEDGES * 4);
  float* sums  = (float*)take((size_t)LDK * 4);
  float* sumsq = (float*)take((size_t)LDK * 4);
  float* counts= (float*)take((size_t)NGRAPHS * 4);
  float* pooled= (float*)take((size_t)NGRAPHS * LDK * 4);
  u16*   pbf   = (u16*)  take((size_t)NGRAPHS * LDK * 2);

  const int T = 256;
  // degrees + normalization coefficients
  zero_f32 <<<cdiv(NNODES, T), T, 0, stream>>>(dinv, NNODES);
  deg_accum<<<cdiv(NEDGES, T), T, 0, stream>>>(col, dinv);
  make_dinv<<<cdiv(NNODES, T), T, 0, stream>>>(dinv);
  make_norm<<<cdiv(NEDGES, T), T, 0, stream>>>(row, col, dinv, norm);
  // initial embeddings
  embed<<<cdiv((long long)NNODES * LDK, T), T, 0, stream>>>(x, emb1, emb2, hbf);

  const int Mtiles = NNODES / 16;          // 6250
  const int nStrips = LDK / 64;            // 5
  for (int l = 0; l < NLAYERS; ++l) {
    convert_wT<<<cdiv(LDK * LDK, T), T, 0, stream>>>(Ws + (size_t)l * EMBD * EMBD, wbf, EMBD, EMBD, LDK);
    int waves = Mtiles * nStrips;
    wmma_gemm<<<cdiv(waves, 8), T, 0, stream>>>(hbf, wbf, bufA, nullptr, Mtiles, nStrips, LDK);
    selfloop_bias<<<cdiv((long long)NNODES * LDK, T), T, 0, stream>>>(bufA, dinv, bs + (size_t)l * EMBD, bufB);
    edge_scatter<<<cdiv((long long)NEDGES * CHUNKS, T), T, 0, stream>>>(row, col, norm, bufA, bufB);
    zero_f32<<<cdiv(LDK, T), T, 0, stream>>>(sums, LDK);
    zero_f32<<<cdiv(LDK, T), T, 0, stream>>>(sumsq, LDK);
    bn_stats<<<512, LDK, 0, stream>>>(bufB, sums, sumsq);
    bn_apply<<<cdiv((long long)NNODES * LDK, T), T, 0, stream>>>(
        bufB, sums, sumsq, gammas + (size_t)l * EMBD, betas + (size_t)l * EMBD,
        hbf, bufA, l != NLAYERS - 1, l == NLAYERS - 1);
  }

  // global mean pool (final BN output is f32 in bufA)
  zero_f32<<<cdiv(NGRAPHS, T), T, 0, stream>>>(counts, NGRAPHS);
  zero_f32<<<cdiv(NGRAPHS * LDK, T), T, 0, stream>>>(pooled, NGRAPHS * LDK);
  count_nodes<<<cdiv(NNODES, T), T, 0, stream>>>(batch, counts);
  pool_acc<<<cdiv((long long)NNODES * CHUNKS, T), T, 0, stream>>>(batch, bufA, pooled);
  pool_div<<<cdiv(NGRAPHS * LDK, T), T, 0, stream>>>(pooled, counts, pbf);

  // final projection: [2048 x 320] @ [320 x 256] + bf
  convert_wT<<<cdiv(FEATD * LDK, T), T, 0, stream>>>(Wf, wfbf, EMBD, FEATD, LDK);
  int fMtiles = NGRAPHS / 16, fStrips = FEATD / 64;
  wmma_gemm<<<cdiv(fMtiles * fStrips, 8), T, 0, stream>>>(pbf, wfbf, out, bfb, fMtiles, fStrips, FEATD);
}
